// THGNN_65317862637865
// MI455X (gfx1250) — matrix operand order repaired and checked by reading, not compiled, not run on hardware
//
#include <hip/hip_runtime.h>
#include <math.h>

#define NN 2048
#define DD 256
#define HH 4
#define FF 64
#define HFD 256
#define TT 64

// Async global->LDS staging (CDNA5 GLOBAL_LOAD_ASYNC_TO_LDS_B128, ASYNCcnt).
// Set to 0 to fall back to VGPR-bounce staging if the mnemonic is rejected.
#define USE_ASYNC_LDS 1

typedef __attribute__((ext_vector_type(16))) _Float16 v16h;
typedef __attribute__((ext_vector_type(8)))  _Float16 v8h;
typedef __attribute__((ext_vector_type(8)))  float    v8f;
typedef unsigned long long ull;

// ---------------- WMMA fragment loaders ----------------
// A fragment: 16x32 f16 tile, lane holds row (lane&15), k-chunks interleaved:
//   halves 0..7  -> k = kk + 8*hi + t
//   halves 8..15 -> k = kk + 16 + 8*hi + (t-8)
__device__ __forceinline__ v16h load_frag_a(const _Float16* base, int ld, int row, int kk) {
    int hi = (threadIdx.x >> 4) & 1;
    const _Float16* p = base + (size_t)row * ld + kk + 8 * hi;
    v8h lo = *(const v8h*)p;
    v8h h8 = *(const v8h*)(p + 16);
    v16h a;
#pragma unroll
    for (int t = 0; t < 8; t++) { a[t] = lo[t]; a[8 + t] = h8[t]; }
    return a;
}
// B fragment from Bt (row-major [Ncols, K] = B transposed): lane holds column
// (lane&15) of B; interleaved k packing: t<8 -> k = kk + 8*hi + t, t>=8 -> k = kk + 16 + 8*hi + (t-8)
__device__ __forceinline__ v16h load_frag_b(const _Float16* baseT, int ld, int col, int kk) {
    int hi = (threadIdx.x >> 4) & 1;
    const _Float16* p = baseT + (size_t)col * ld + kk + 8 * hi;
    v8h lo = *(const v8h*)p;
    v8h h8 = *(const v8h*)(p + 16);
    v16h b;
#pragma unroll
    for (int t = 0; t < 8; t++) { b[t] = lo[t]; b[8 + t] = h8[t]; }
    return b;
}

// ---------------- small prep kernels ----------------
__global__ void prep_yc_kernel(const float* __restrict__ y, _Float16* __restrict__ yc,
                               float* __restrict__ rstd) {
    __shared__ float red[TT];
    int n = blockIdx.x, t = threadIdx.x;           // 64 threads
    float v = y[n * TT + t];
    red[t] = v; __syncthreads();
    for (int s = 32; s > 0; s >>= 1) { if (t < s) red[t] += red[t + s]; __syncthreads(); }
    float mean = red[0] / (float)TT;
    __syncthreads();
    float d = v - mean;
    yc[n * TT + t] = (_Float16)d;
    red[t] = d * d; __syncthreads();
    for (int s = 32; s > 0; s >>= 1) { if (t < s) red[t] += red[t + s]; __syncthreads(); }
    if (t == 0) rstd[n] = rsqrtf(red[0] / (float)(TT - 1) + 1e-30f);
}

__global__ void cvt_f16_kernel(const float* __restrict__ in, _Float16* __restrict__ out, int n) {
    int i = blockIdx.x * blockDim.x + threadIdx.x;
    if (i < n) out[i] = (_Float16)in[i];
}

// transpose + convert: in f32 [R,C] -> out f16 [C,R]
__global__ void transpose_f16_kernel(const float* __restrict__ in, _Float16* __restrict__ out,
                                     int R, int C) {
    int idx = blockIdx.x * blockDim.x + threadIdx.x;
    if (idx < R * C) {
        int r = idx / C, c = idx % C;
        out[(size_t)c * R + r] = (_Float16)in[(size_t)r * C + c];
    }
}

// ---------------- corr + mask (WMMA on yc) ----------------
__global__ void corr_mask_kernel(const _Float16* __restrict__ yc, const float* __restrict__ rstd,
                                 const float* __restrict__ adj, unsigned char* __restrict__ mask) {
    int lane = threadIdx.x, l16 = lane & 15, hi = lane >> 4;
    int i0 = blockIdx.y * 16, j0 = blockIdx.x * 16;
    v8f c = {};
#pragma unroll
    for (int kk = 0; kk < TT; kk += 32) {
        v16h a = load_frag_a(yc, TT, i0 + l16, kk);  // rows i
        v16h b = load_frag_b(yc, TT, j0 + l16, kk);  // B = yc^T so B^T rows = yc rows j
        c = __builtin_amdgcn_wmma_f32_16x16x32_f16(false, a, false, b, (short)0, c, false, false);
    }
    int j = j0 + l16;
    float rj = rstd[j];
#pragma unroll
    for (int v = 0; v < 8; v++) {
        int i = i0 + v + 8 * hi;
        float corr = c[v] * rstd[i] * rj;
        float av = adj[(size_t)i * NN + j];
        unsigned char code = (av != 0.0f) ? (corr >= 0.1f ? 1 : 2) : 0;
        mask[(size_t)i * NN + j] = code;
    }
}

// ---------------- 4-wave WMMA GEMM, 64x64 macro-tile, B staged in LDS ----------------
// C[M,Nc] = A[M,K] @ B  with Bt = [Nc,K] (B transposed, f16 row-major).
// Block = 128 threads (4 waves). Wave w owns rows [m0+16w, m0+16w+16).
// Per k-step the 64col x 32k B panel is async-copied to LDS and shared.
#define BROW 40   // padded LDS row stride in halves (bank-conflict-free)
__global__ void wmma_gemm_kernel(const _Float16* __restrict__ A, const _Float16* __restrict__ Bt,
                                 float* __restrict__ C, _Float16* __restrict__ C16,
                                 const float* __restrict__ bias, int M, int Nc, int K, int act) {
    __shared__ _Float16 lds_b[64 * BROW];
    int tid = threadIdx.x;
    int lane = tid & 31, wave = tid >> 5;
    int l16 = lane & 15, hi = (lane >> 4) & 1;
    int n0 = blockIdx.x * 64;
    int m0 = blockIdx.y * 64 + wave * 16;
    int arow = m0 + l16;
    v8f acc[4] = {};
    for (int kk = 0; kk < K; kk += 32) {
        __syncthreads();   // LDS reuse across k-steps
#pragma unroll
        for (int it = 0; it < 2; it++) {
            int f = tid * 8 + it * 1024;          // flat half index, 2048 total
            int col = f >> 5, k = f & 31;
            const _Float16* g = Bt + (size_t)(n0 + col) * K + kk + k;
            _Float16* l = lds_b + col * BROW + k;
#if USE_ASYNC_LDS
            unsigned lofs = (unsigned)(size_t)l;
            asm volatile("global_load_async_to_lds_b128 %0, %1, off"
                         :: "v"(lofs), "v"(g) : "memory");
#else
            *(v8h*)l = *(const v8h*)g;
#endif
        }
#if USE_ASYNC_LDS
        asm volatile("s_wait_asynccnt 0x0" ::: "memory");
#endif
        __syncthreads();
        v16h a = load_frag_a(A, K, arow, kk);
#pragma unroll
        for (int u = 0; u < 4; u++) {
            const _Float16* p = lds_b + (u * 16 + l16) * BROW + 8 * hi;
            v8h lo = *(const v8h*)p;
            v8h h8 = *(const v8h*)(p + 16);
            v16h b;
#pragma unroll
            for (int t = 0; t < 8; t++) { b[t] = lo[t]; b[8 + t] = h8[t]; }
            acc[u] = __builtin_amdgcn_wmma_f32_16x16x32_f16(false, a, false, b, (short)0,
                                                            acc[u], false, false);
        }
    }
#pragma unroll
    for (int u = 0; u < 4; u++) {
        int n = n0 + u * 16 + l16;
        float bv = bias ? bias[n] : 0.0f;
#pragma unroll
        for (int v = 0; v < 8; v++) {
            int m = m0 + v + 8 * hi;
            float val = acc[u][v] + bv;
            if (act == 1) val = tanhf(val);
            C[(size_t)m * Nc + n] = val;
            if (C16) C16[(size_t)m * Nc + n] = (_Float16)val;
        }
    }
}

// ---------------- per-(head,node) attention scores ----------------
__global__ void f1f2_kernel(const float* __restrict__ s, const float* __restrict__ wu,
                            const float* __restrict__ wv, float* __restrict__ f1,
                            float* __restrict__ f2) {
    int idx = blockIdx.x * blockDim.x + threadIdx.x;
    if (idx >= HH * NN) return;
    int h = idx / NN, n = idx % NN;
    const float* sp = s + (size_t)n * HFD + h * FF;
    float a = 0.f, b = 0.f;
#pragma unroll 4
    for (int f = 0; f < FF; f++) { float sv = sp[f]; a += sv * wu[h * FF + f]; b += sv * wv[h * FF + f]; }
    f1[idx] = a; f2[idx] = b;
}

// ---------------- masked online-softmax stats, one wave per (i,h) ----------------
__global__ void attn_stats_kernel(const float* __restrict__ f1, const float* __restrict__ f2,
                                  const unsigned char* __restrict__ mask, int code,
                                  float* __restrict__ rowmax, float* __restrict__ rdenom) {
    int lane = threadIdx.x;
    int i = blockIdx.x, h = blockIdx.y;
    float f2i = f2[h * NN + i];
    const unsigned char* mp = mask + (size_t)i * NN;
    float m = -1e30f, s = 0.f;
    for (int j = lane; j < NN; j += 32) {
        if ((int)mp[j] == code) {
            float lg = f1[h * NN + j] + f2i;
            lg = lg > 0.f ? lg : 0.2f * lg;
            if (lg > m) { s = s * __expf(m - lg) + 1.f; m = lg; }
            else         { s += __expf(lg - m); }
        }
    }
#pragma unroll
    for (int off = 16; off > 0; off >>= 1) {
        float m2 = __shfl_xor(m, off, 32);
        float s2 = __shfl_xor(s, off, 32);
        float mm = fmaxf(m, m2);
        s = s * __expf(m - mm) + s2 * __expf(m2 - mm);
        m = mm;
    }
    if (lane == 0) {
        rowmax[h * NN + i] = m;
        rdenom[h * NN + i] = (s > 0.f) ? (1.f / s) : 0.f;
    }
}

// ---------------- attn @ s via WMMA ----------------
// One wave per (i-tile, head); probs built once per j-chunk and fed to 4
// back-to-back WMMAs (all F=64 columns). Per-lane mask/f1 data are two
// contiguous 8-element spans -> vector loads instead of 32 scalar loads.
__global__ void attn_out_kernel(const float* __restrict__ f1, const float* __restrict__ f2,
                                const unsigned char* __restrict__ mask, int code,
                                const float* __restrict__ rowmax, const float* __restrict__ rdenom,
                                const _Float16* __restrict__ sT, float* __restrict__ outb) {
    int lane = threadIdx.x, l16 = lane & 15, hi = lane >> 4;
    int i0 = blockIdx.x * 16;
    int h  = blockIdx.y;
    int i = i0 + l16;
    float f2i  = f2[h * NN + i];
    float mrow = rowmax[h * NN + i];
    float rd   = rdenom[h * NN + i];
    const unsigned char* mp = mask + (size_t)i * NN;
    const float* f1p = f1 + h * NN;
    v8f c[4] = {};
    for (int j0 = 0; j0 < NN; j0 += 32) {
        ull mq0 = *(const ull*)(mp + j0 + 8 * hi);            // 8 mask bytes, span 0
        ull mq1 = *(const ull*)(mp + j0 + 16 + 8 * hi);       // 8 mask bytes, span 1
        v8f F0  = *(const v8f*)(f1p + j0 + 8 * hi);           // 8 f1 values, span 0
        v8f F1  = *(const v8f*)(f1p + j0 + 16 + 8 * hi);      // 8 f1 values, span 1
        v16h a;
#pragma unroll
        for (int t = 0; t < 8; t++) {
            float p0 = 0.f, p1 = 0.f;
            if ((int)((mq0 >> (8 * t)) & 0xFF) == code) {
                float lg = F0[t] + f2i;
                lg = lg > 0.f ? lg : 0.2f * lg;
                p0 = __expf(lg - mrow) * rd;
            }
            if ((int)((mq1 >> (8 * t)) & 0xFF) == code) {
                float lg = F1[t] + f2i;
                lg = lg > 0.f ? lg : 0.2f * lg;
                p1 = __expf(lg - mrow) * rd;
            }
            a[t] = (_Float16)p0;
            a[8 + t] = (_Float16)p1;
        }
#pragma unroll
        for (int u = 0; u < 4; u++) {
            v16h b = load_frag_b(sT, NN, h * FF + u * 16 + l16, j0);
            c[u] = __builtin_amdgcn_wmma_f32_16x16x32_f16(false, a, false, b, (short)0,
                                                          c[u], false, false);
        }
    }
#pragma unroll
    for (int u = 0; u < 4; u++) {
#pragma unroll
        for (int v = 0; v < 8; v++) {
            int m = i0 + v + 8 * hi;
            outb[(size_t)m * HFD + h * FF + u * 16 + l16] = c[u][v];
        }
    }
}

// ---------------- GAT epilogue: out = attn_out + b + emb@pw + pb -> f16 ----------------
__global__ void gat_fuse_kernel(const float* __restrict__ attn_out, const float* __restrict__ proj,
                                const float* __restrict__ b, const float* __restrict__ pb,
                                _Float16* __restrict__ out16) {
    int idx = blockIdx.x * blockDim.x + threadIdx.x;
    if (idx < NN * HFD) {
        int col = idx % HFD;
        out16[idx] = (_Float16)(attn_out[idx] + proj[idx] + b[col] + pb[col]);
    }
}

// ---------------- semantic attention tail ----------------
__global__ void wlog_kernel(const float* __restrict__ t, const float* __restrict__ w2,
                            float* __restrict__ wlog) {
    int r = blockIdx.x * blockDim.x + threadIdx.x;
    if (r < 3 * NN) {
        const float* tp = t + (size_t)r * DD;
        float a = 0.f;
#pragma unroll 4
        for (int d = 0; d < DD; d++) a += tp[d] * w2[d];
        wlog[r] = a;
    }
}

__global__ void combine_kernel(const float* __restrict__ ch0, const float* __restrict__ ch1,
                               const float* __restrict__ ch2, const float* __restrict__ wlog,
                               float* __restrict__ x) {
    int idx = blockIdx.x * blockDim.x + threadIdx.x;
    if (idx < NN * DD) {
        int n = idx / DD;
        float w0 = wlog[n], w1 = wlog[NN + n], w2 = wlog[2 * NN + n];
        float mx = fmaxf(w0, fmaxf(w1, w2));
        float e0 = __expf(w0 - mx), e1 = __expf(w1 - mx), e2 = __expf(w2 - mx);
        float inv = 1.f / (e0 + e1 + e2);
        x[idx] = (e0 * ch0[idx] + e1 * ch1[idx] + e2 * ch2[idx]) * inv;
    }
}

__global__ void colmean_kernel(const float* __restrict__ x, float* __restrict__ mean) {
    __shared__ float red[256];
    int d = blockIdx.x, t = threadIdx.x;
    float a = 0.f;
    for (int n = t; n < NN; n += 256) a += x[(size_t)n * DD + d];
    red[t] = a; __syncthreads();
    for (int s = 128; s > 0; s >>= 1) { if (t < s) red[t] += red[t + s]; __syncthreads(); }
    if (t == 0) mean[d] = red[0] / (float)NN;
}

__global__ void pairnorm_kernel(const float* __restrict__ x, const float* __restrict__ mean,
                                float* __restrict__ out) {
    __shared__ float red[DD];
    int n = blockIdx.x, d = threadIdx.x;   // 256 threads
    float v = x[(size_t)n * DD + d] - mean[d];
    red[d] = v * v; __syncthreads();
    for (int s = 128; s > 0; s >>= 1) { if (d < s) red[d] += red[d + s]; __syncthreads(); }
    out[(size_t)n * DD + d] = v / sqrtf(1e-6f + red[0]);
}

// ---------------- host side ----------------
static size_t g_off;
static inline void* bump(char* base, size_t bytes) {
    void* p = base + g_off;
    g_off = (g_off + bytes + 255) & ~(size_t)255;
    return p;
}

extern "C" void kernel_launch(void* const* d_in, const int* in_sizes, int n_in,
                              void* d_out, int out_size, void* d_ws, size_t ws_size,
                              hipStream_t stream) {
    const float* emb      = (const float*)d_in[0];
    const float* y_as_x   = (const float*)d_in[1];
    const float* adj      = (const float*)d_in[2];
    const float* pos_w    = (const float*)d_in[3];
    const float* pos_wu   = (const float*)d_in[4];
    const float* pos_wv   = (const float*)d_in[5];
    const float* pos_b    = (const float*)d_in[6];
    const float* pos_pw   = (const float*)d_in[7];
    const float* pos_pb   = (const float*)d_in[8];
    const float* neg_w    = (const float*)d_in[9];
    const float* neg_wu   = (const float*)d_in[10];
    const float* neg_wv   = (const float*)d_in[11];
    const float* neg_b    = (const float*)d_in[12];
    const float* neg_pw   = (const float*)d_in[13];
    const float* neg_pb   = (const float*)d_in[14];
    const float* self_w   = (const float*)d_in[15];
    const float* self_b   = (const float*)d_in[16];
    const float* mlp_pos_w= (const float*)d_in[17];
    const float* mlp_pos_b= (const float*)d_in[18];
    const float* mlp_neg_w= (const float*)d_in[19];
    const float* mlp_neg_b= (const float*)d_in[20];
    const float* sem_w1   = (const float*)d_in[21];
    const float* sem_b1   = (const float*)d_in[22];
    const float* sem_w2   = (const float*)d_in[23];
    float* outp = (float*)d_out;

    char* ws = (char*)d_ws;
    g_off = 0;
    _Float16* yc16   = (_Float16*)bump(ws, (size_t)NN * TT * 2);
    float*    rstd   = (float*)   bump(ws, (size_t)NN * 4);
    _Float16* emb16  = (_Float16*)bump(ws, (size_t)NN * DD * 2);
    unsigned char* mask = (unsigned char*)bump(ws, (size_t)NN * NN);
    _Float16* poswT  = (_Float16*)bump(ws, (size_t)DD * HFD * 2);
    _Float16* negwT  = (_Float16*)bump(ws, (size_t)DD * HFD * 2);
    _Float16* pospwT = (_Float16*)bump(ws, (size_t)DD * HFD * 2);
    _Float16* negpwT = (_Float16*)bump(ws, (size_t)DD * HFD * 2);
    _Float16* selfwT = (_Float16*)bump(ws, (size_t)DD * DD * 2);
    _Float16* mlpposT= (_Float16*)bump(ws, (size_t)HFD * DD * 2);
    _Float16* mlpnegT= (_Float16*)bump(ws, (size_t)HFD * DD * 2);
    _Float16* semw1T = (_Float16*)bump(ws, (size_t)DD * DD * 2);
    float* s_pos     = (float*)   bump(ws, (size_t)NN * HFD * 4);
    float* s_neg     = (float*)   bump(ws, (size_t)NN * HFD * 4);
    _Float16* sposT  = (_Float16*)bump(ws, (size_t)HFD * NN * 2);
    _Float16* snegT  = (_Float16*)bump(ws, (size_t)HFD * NN * 2);
    float* proj_pos  = (float*)   bump(ws, (size_t)NN * HFD * 4);
    float* proj_neg  = (float*)   bump(ws, (size_t)NN * HFD * 4);
    float* f1p = (float*)bump(ws, (size_t)HH * NN * 4);
    float* f2p = (float*)bump(ws, (size_t)HH * NN * 4);
    float* f1n = (float*)bump(ws, (size_t)HH * NN * 4);
    float* f2n = (float*)bump(ws, (size_t)HH * NN * 4);
    float* rmaxp = (float*)bump(ws, (size_t)HH * NN * 4);
    float* rdenp = (float*)bump(ws, (size_t)HH * NN * 4);
    float* rmaxn = (float*)bump(ws, (size_t)HH * NN * 4);
    float* rdenn = (float*)bump(ws, (size_t)HH * NN * 4);
    float* aout_p = (float*)bump(ws, (size_t)NN * HFD * 4);
    float* aout_n = (float*)bump(ws, (size_t)NN * HFD * 4);
    _Float16* gatp16 = (_Float16*)bump(ws, (size_t)NN * HFD * 2);
    _Float16* gatn16 = (_Float16*)bump(ws, (size_t)NN * HFD * 2);
    float* ch0 = (float*)bump(ws, (size_t)NN * DD * 4);
    float* ch1 = (float*)bump(ws, (size_t)NN * DD * 4);
    float* ch2 = (float*)bump(ws, (size_t)NN * DD * 4);
    _Float16* allemb16 = (_Float16*)bump(ws, (size_t)3 * NN * DD * 2);
    float* t_f32 = (float*)bump(ws, (size_t)3 * NN * DD * 4);
    float* wlog  = (float*)bump(ws, (size_t)3 * NN * 4);
    float* xbuf  = (float*)bump(ws, (size_t)NN * DD * 4);
    float* cmean = (float*)bump(ws, (size_t)DD * 4);
    (void)ws_size; (void)n_in; (void)in_sizes; (void)out_size;

    const int TPB = 256;
    // 1) correlation prep
    prep_yc_kernel<<<NN, TT, 0, stream>>>(y_as_x, yc16, rstd);
    // 2) f16 copies / transposed weights
    cvt_f16_kernel<<<(NN * DD + TPB - 1) / TPB, TPB, 0, stream>>>(emb, emb16, NN * DD);
    transpose_f16_kernel<<<(DD * HFD + TPB - 1) / TPB, TPB, 0, stream>>>(pos_w, poswT, DD, HFD);
    transpose_f16_kernel<<<(DD * HFD + TPB - 1) / TPB, TPB, 0, stream>>>(neg_w, negwT, DD, HFD);
    transpose_f16_kernel<<<(DD * HFD + TPB - 1) / TPB, TPB, 0, stream>>>(pos_pw, pospwT, DD, HFD);
    transpose_f16_kernel<<<(DD * HFD + TPB - 1) / TPB, TPB, 0, stream>>>(neg_pw, negpwT, DD, HFD);
    transpose_f16_kernel<<<(DD * DD + TPB - 1) / TPB, TPB, 0, stream>>>(self_w, selfwT, DD, DD);
    transpose_f16_kernel<<<(HFD * DD + TPB - 1) / TPB, TPB, 0, stream>>>(mlp_pos_w, mlpposT, HFD, DD);
    transpose_f16_kernel<<<(HFD * DD + TPB - 1) / TPB, TPB, 0, stream>>>(mlp_neg_w, mlpnegT, HFD, DD);
    transpose_f16_kernel<<<(DD * DD + TPB - 1) / TPB, TPB, 0, stream>>>(sem_w1, semw1T, DD, DD);
    // 3) corr + edge classification (WMMA)
    corr_mask_kernel<<<dim3(NN / 16, NN / 16), 32, 0, stream>>>(yc16, rstd, adj, mask);
    // 4) s = emb @ w  (WMMA, 64x64 macro-tile), then s^T f16 for the attention GEMM
    wmma_gemm_kernel<<<dim3(HFD / 64, NN / 64), 128, 0, stream>>>(emb16, poswT, s_pos, nullptr, nullptr, NN, HFD, DD, 0);
    wmma_gemm_kernel<<<dim3(HFD / 64, NN / 64), 128, 0, stream>>>(emb16, negwT, s_neg, nullptr, nullptr, NN, HFD, DD, 0);
    transpose_f16_kernel<<<(NN * HFD + TPB - 1) / TPB, TPB, 0, stream>>>(s_pos, sposT, NN, HFD);
    transpose_f16_kernel<<<(NN * HFD + TPB - 1) / TPB, TPB, 0, stream>>>(s_neg, snegT, NN, HFD);
    // 5) residual projections emb @ pw (WMMA)
    wmma_gemm_kernel<<<dim3(HFD / 64, NN / 64), 128, 0, stream>>>(emb16, pospwT, proj_pos, nullptr, nullptr, NN, HFD, DD, 0);
    wmma_gemm_kernel<<<dim3(HFD / 64, NN / 64), 128, 0, stream>>>(emb16, negpwT, proj_neg, nullptr, nullptr, NN, HFD, DD, 0);
    // 6) sup = emb @ self_w + self_b (also f16 channel 0)
    wmma_gemm_kernel<<<dim3(DD / 64, NN / 64), 128, 0, stream>>>(emb16, selfwT, ch0, allemb16, self_b, NN, DD, DD, 0);
    // 7) attention scores + online softmax stats
    f1f2_kernel<<<(HH * NN + TPB - 1) / TPB, TPB, 0, stream>>>(s_pos, pos_wu, pos_wv, f1p, f2p);
    f1f2_kernel<<<(HH * NN + TPB - 1) / TPB, TPB, 0, stream>>>(s_neg, neg_wu, neg_wv, f1n, f2n);
    attn_stats_kernel<<<dim3(NN, HH), 32, 0, stream>>>(f1p, f2p, mask, 1, rmaxp, rdenp);
    attn_stats_kernel<<<dim3(NN, HH), 32, 0, stream>>>(f1n, f2n, mask, 2, rmaxn, rdenn);
    // 8) attn @ s (WMMA, probs built once per chunk, 4 WMMAs back-to-back)
    attn_out_kernel<<<dim3(NN / 16, HH), 32, 0, stream>>>(f1p, f2p, mask, 1, rmaxp, rdenp, sposT, aout_p);
    attn_out_kernel<<<dim3(NN / 16, HH), 32, 0, stream>>>(f1n, f2n, mask, 2, rmaxn, rdenn, snegT, aout_n);
    // 9) GAT epilogues -> f16 inputs for MLP GEMMs
    gat_fuse_kernel<<<(NN * HFD + TPB - 1) / TPB, TPB, 0, stream>>>(aout_p, proj_pos, pos_b, pos_pb, gatp16);
    gat_fuse_kernel<<<(NN * HFD + TPB - 1) / TPB, TPB, 0, stream>>>(aout_n, proj_neg, neg_b, neg_pb, gatn16);
    // 10) relation MLPs (WMMA) -> channels 1,2 (f32 + f16)
    wmma_gemm_kernel<<<dim3(DD / 64, NN / 64), 128, 0, stream>>>(gatp16, mlpposT, ch1, allemb16 + (size_t)NN * DD, mlp_pos_b, NN, DD, HFD, 0);
    wmma_gemm_kernel<<<dim3(DD / 64, NN / 64), 128, 0, stream>>>(gatn16, mlpnegT, ch2, allemb16 + (size_t)2 * NN * DD, mlp_neg_b, NN, DD, HFD, 0);
    // 11) semantic attention: t = tanh(all_emb @ sem_w1 + b1) (WMMA, M = 3N)
    wmma_gemm_kernel<<<dim3(DD / 64, 3 * NN / 64), 128, 0, stream>>>(allemb16, semw1T, t_f32, nullptr, sem_b1, 3 * NN, DD, DD, 1);
    wlog_kernel<<<(3 * NN + TPB - 1) / TPB, TPB, 0, stream>>>(t_f32, sem_w2, wlog);
    combine_kernel<<<(NN * DD + TPB - 1) / TPB, TPB, 0, stream>>>(ch0, ch1, ch2, wlog, xbuf);
    // 12) PairNorm PN-SI
    colmean_kernel<<<DD, 256, 0, stream>>>(xbuf, cmean);
    pairnorm_kernel<<<NN, DD, 0, stream>>>(xbuf, cmean, outp);
}
